// MCorrLCorr_73821897884184
// MI455X (gfx1250) — matrix operand order repaired and checked by reading, compiled, not run on hardware
//
#include <hip/hip_runtime.h>

// MCorrLCorr as implicit GEMM: out[a,b,n,o] = sum_{w1,w2,c} in[(a+1)(w1+1)-1, b+w2, n, c]*W[w1,w2,c,o] + bias
// M=48576, N=256, K=2048.  ~51 GFLOP vs ~150MB HBM -> memory bound with bf16 WMMA.
// Weights pre-converted to bf16 in d_ws (once); B chunks staged with
// GLOBAL_LOAD_ASYNC_TO_LDS_B128 (ASYNCcnt).  Each wave computes a 2(b) x 4(nt)
// tile group: 8 WMMAs per 4 A ds_load_b128 + 64 16-bit B LDS loads (vs 18 before), B frags reused x2.

typedef __bf16         v16bf  __attribute__((ext_vector_type(16)));
typedef float          v8f    __attribute__((ext_vector_type(8)));
typedef unsigned short u16x8  __attribute__((ext_vector_type(8)));
typedef unsigned short u16x16 __attribute__((ext_vector_type(16)));

#define DX1   48
#define DX2   256
#define DN    16
#define DCIN  128
#define DCOUT 256
#define DY1   12
#define DY2   253
#define BT    4

// LDS geometry (bf16 elements); rows padded for bank skew.
#define A_ROW   136
#define A_BLK   (16 * A_ROW)          // 2176
#define NABLK   7
#define B_ROW   264
#define B_CHUNK (32 * B_ROW)          // 8448
#define B_OFF   (NABLK * A_BLK)       // 15232
#define LDS_ELS (B_OFF + 2 * B_CHUNK) // 64256 B total

#define HAS_ASYNC_LDS __has_builtin(__builtin_amdgcn_global_load_async_to_lds_b128)

__device__ __forceinline__ unsigned short f2bf(float f) {
  unsigned int u = __builtin_bit_cast(unsigned int, f);
  u += 0x7FFFu + ((u >> 16) & 1u);                 // RNE
  return (unsigned short)(u >> 16);
}

__device__ __forceinline__ unsigned int pk2(float x, float y) {
#if __has_builtin(__builtin_amdgcn_cvt_pk_bf16_f32)
  typedef __bf16 v2bf __attribute__((ext_vector_type(2)));
  v2bf v = __builtin_amdgcn_cvt_pk_bf16_f32(x, y);
  return __builtin_bit_cast(unsigned int, v);
#else
  return (unsigned int)f2bf(x) | ((unsigned int)f2bf(y) << 16);
#endif
}

#if HAS_ASYNC_LDS
// builtin signature (from hipcc diagnostic): (int4 AS1*, int4 AS3*, imm, imm)
typedef int v4i_vs __attribute__((vector_size(16)));
typedef __attribute__((address_space(1))) v4i_vs* g4p;
typedef __attribute__((address_space(3))) v4i_vs* l4p;
__device__ __forceinline__ void async_cp16(const unsigned short* g, unsigned short* l) {
  __builtin_amdgcn_global_load_async_to_lds_b128((g4p)g, (l4p)l, 0, 0);
}
#endif

__device__ __forceinline__ void wait_async0() {
#if HAS_ASYNC_LDS
#if __has_builtin(__builtin_amdgcn_s_wait_asynccnt)
  __builtin_amdgcn_s_wait_asynccnt(0);
#else
  asm volatile("s_wait_asynccnt 0x0" ::: "memory");
#endif
#endif
}

// one-shot weight convert: f32 [4,4,128,256] -> bf16 flat, RNE
__global__ __launch_bounds__(256)
void MCorr_wprep_kernel(const float* __restrict__ w, unsigned short* __restrict__ o) {
  const int idx = blockIdx.x * 256 + threadIdx.x;   // float4 units, 0..131071
  float4 v = *(const float4*)(w + (size_t)idx * 4);
  uint2 p;
  p.x = pk2(v.x, v.y);
  p.y = pk2(v.z, v.w);
  *(uint2*)(o + (size_t)idx * 4) = p;
}

__global__ __launch_bounds__(256)
void MCorrLCorr_73821897884184_kernel(const float* __restrict__ in_,
                                      const float* __restrict__ wgt,
                                      const float* __restrict__ bias,
                                      const unsigned short* __restrict__ wbf, // bf16 weights or null
                                      float* __restrict__ out) {
  __shared__ unsigned short lds[LDS_ELS];

  const int tid   = threadIdx.x;
  const int lane  = tid & 31;
  const int wave  = tid >> 5;
  const int a     = blockIdx.y;
  const int b0    = blockIdx.x * BT;
  const int bpair = wave & 1;                // b-values {bpair*2, bpair*2+1}
  const int oquad = wave >> 1;               // 64-wide cout slice
  const int o0    = oquad * 64;

  auto stageB = [&](int chunk, int buf) {
    unsigned short* dst = lds + B_OFF + buf * B_CHUNK;
#if HAS_ASYNC_LDS
    if (wbf) {
      const unsigned short* src = wbf + (size_t)chunk * (32 * 256);
#pragma unroll
      for (int i = 0; i < 4; ++i) {
        int idx = tid + i * 256;            // 16B segments
        int r   = idx >> 5;
        int o8  = (idx & 31) << 3;
        async_cp16(src + r * 256 + o8, dst + r * B_ROW + o8);
      }
      return;
    }
#endif
    const float* src = wgt + (size_t)chunk * (32 * 256);
#pragma unroll 2
    for (int i = 0; i < 8; ++i) {
      int idx = tid + i * 256;
      int r   = idx >> 6;
      int o4  = (idx & 63) << 2;
      float4 v = *(const float4*)(src + r * 256 + o4);
      uint2 p; p.x = pk2(v.x, v.y); p.y = pk2(v.z, v.w);
      *(uint2*)(dst + r * B_ROW + o4) = p;
    }
  };

  v8f acc[8];                                // [bb*4 + nt]
#pragma unroll
  for (int i = 0; i < 8; ++i)
#pragma unroll
    for (int j = 0; j < 8; ++j) acc[i][j] = 0.f;

  int cur = 0;
  stageB(0, 0);

  for (int step = 0; step < 64; ++step) {
    const int w1 = step >> 4;
    const int w2 = (step >> 2) & 3;
    const int kc = step & 3;

    if ((step & 15) == 0) {
      wait_async0();
      __syncthreads();                       // old A consumed by all waves
      const int x1 = (a + 1) * (w1 + 1) - 1;
#pragma unroll 2
      for (int i = 0; i < 14; ++i) {
        int idx = tid + i * 256;
        int j   = idx >> 9;
        int q   = idx & 511;
        int n   = q >> 5;
        int c   = (q & 31) << 2;
        int x2  = b0 + j; if (x2 > DX2 - 1) x2 = DX2 - 1;
        float4 v = *(const float4*)(in_ +
                     (((size_t)x1 * DX2 + x2) * DN + n) * DCIN + c);
        uint2 p; p.x = pk2(v.x, v.y); p.y = pk2(v.z, v.w);
        *(uint2*)(lds + j * A_BLK + n * A_ROW + c) = p;
      }
    }
    wait_async0();                           // my async B writes done
    __syncthreads();                         // => everyone's B[cur]/A visible

    if (step + 1 < 64) stageB(step + 1, cur ^ 1);
    if (step + 2 < 64) {
      const void* pf = wbf
          ? (const void*)(wbf + (size_t)(step + 2) * (32 * 256) + (tid << 4))
          : (const void*)(wgt + (size_t)(step + 2) * (32 * 256) + (tid << 5));
      __builtin_prefetch(pf, 0, 1);
    }

    // ---- compute: 2 A frags + 4 B frags -> 8 WMMAs (4 ds_load_b128 + 64 16-bit B loads) ----
    const int ab0 = bpair * 2 + w2;          // A blocks ab0, ab0+1
    const unsigned short* ap = lds + ab0 * A_BLK + (lane & 15) * A_ROW
                             + kc * 32 + ((lane >> 4) << 3);
    v16bf afrag[2];
#pragma unroll
    for (int bb = 0; bb < 2; ++bb) {
      u16x8 alo = *(const u16x8*)(ap + bb * A_BLK);
      u16x8 ahi = *(const u16x8*)(ap + bb * A_BLK + 16);
      u16x16 au;
#pragma unroll
      for (int i = 0; i < 8; ++i) { au[i] = alo[i]; au[8 + i] = ahi[i]; }
      afrag[bb] = __builtin_bit_cast(v16bf, au);
    }

    const unsigned short* bp = lds + B_OFF + cur * B_CHUNK + ((lane >> 4) << 3) * B_ROW + o0 + (lane & 15);
    u16x8 blo[4], bhi[4];
#pragma unroll
    for (int nt = 0; nt < 4; ++nt) {
#pragma unroll
      for (int i = 0; i < 8; ++i) {
        blo[nt][i] = bp[i * B_ROW + nt * 16];
        bhi[nt][i] = bp[(16 + i) * B_ROW + nt * 16];
      }
    }
#pragma unroll
    for (int nt = 0; nt < 4; ++nt) {
      u16x16 bu;
#pragma unroll
      for (int i = 0; i < 8; ++i) { bu[i] = blo[nt][i]; bu[8 + i] = bhi[nt][i]; }
      const v16bf bfrag = __builtin_bit_cast(v16bf, bu);
#pragma unroll
      for (int bb = 0; bb < 2; ++bb) {       // B frag reused from registers
        acc[bb * 4 + nt] = __builtin_amdgcn_wmma_f32_16x16x32_bf16(
            false, afrag[bb], false, bfrag, (short)0, acc[bb * 4 + nt], false, false);
      }
    }
    cur ^= 1;
  }

  // ---- epilogue ----
  const int col    = lane & 15;
  const int rowoff = (lane >> 4) << 3;
#pragma unroll
  for (int bb = 0; bb < 2; ++bb) {
    const int b = b0 + bpair * 2 + bb;
    if (b >= DY2) continue;
    const size_t base0 = (((size_t)a * DY2 + b) * DN) * DCOUT;
#pragma unroll
    for (int nt = 0; nt < 4; ++nt) {
      const int o  = o0 + nt * 16 + col;
      const float bv = bias[o];
#pragma unroll
      for (int r = 0; r < 8; ++r) {
        const int n = r + rowoff;
        out[base0 + (size_t)n * DCOUT + o] = acc[bb * 4 + nt][r] + bv;
      }
    }
  }
}

extern "C" void kernel_launch(void* const* d_in, const int* in_sizes, int n_in,
                              void* d_out, int out_size, void* d_ws, size_t ws_size,
                              hipStream_t stream) {
  const float* in_  = (const float*)d_in[0];   // (48,256,16,128) f32
  const float* wgt  = (const float*)d_in[1];   // (4,4,128,256)   f32
  const float* bias = (const float*)d_in[2];   // (256,)          f32
  float* out = (float*)d_out;                  // (12,253,16,256) f32

  const size_t wbytes = (size_t)4 * 4 * 128 * 256 * sizeof(unsigned short); // 1 MB
  const unsigned short* wbf = nullptr;
  if (ws_size >= wbytes) {
    MCorr_wprep_kernel<<<512, 256, 0, stream>>>(wgt, (unsigned short*)d_ws);
    wbf = (const unsigned short*)d_ws;
  }

  dim3 grid((DY2 + BT - 1) / BT, DY1, 1);      // 64 x 12
  MCorrLCorr_73821897884184_kernel<<<grid, 256, 0, stream>>>(in_, wgt, bias, wbf, out);
}
